// MultiHeadAttention_78108275245117
// MI455X (gfx1250) — matrix operand order repaired and checked by reading, not compiled, not run on hardware
//
#include <hip/hip_runtime.h>
#include <hip/hip_bf16.h>

// MI455X (gfx1250) multi-head attention, bf16 WMMA path (f32 accumulate).
// B=2, S=2048, d_model=1024, H=16, d_head=64.

#define S_LEN   2048
#define DMODEL  1024
#define NHEADS  16
#define DHEAD   64
#define NBATCH  2

typedef __attribute__((ext_vector_type(16))) __bf16 v16bf;
typedef __attribute__((ext_vector_type(8)))  __bf16 v8bf;
typedef __attribute__((ext_vector_type(8)))  float  v8f;
typedef unsigned short ushort_t;

// Native bf16 conversion: let the backend pick packed cvt ops.
__device__ __forceinline__ ushort_t f2bf_bits(float f) {
  __bf16 h = (__bf16)f;
  ushort_t u;
  __builtin_memcpy(&u, &h, 2);
  return u;
}

// Pack two floats into one dword of 2x bf16 (encourages v_cvt_pk selection).
__device__ __forceinline__ unsigned pk2bf(float lo, float hi) {
  return (unsigned)f2bf_bits(lo) | ((unsigned)f2bf_bits(hi) << 16);
}

__device__ __forceinline__ v16bf cat8(v8bf lo, v8bf hi) {
  return __builtin_shufflevector(lo, hi, 0,1,2,3,4,5,6,7,8,9,10,11,12,13,14,15);
}

__device__ __forceinline__ v8f wmma_bf16(v16bf a, v16bf b, v8f c) {
  // D = A(16x32) * B(32x16) + C, f32 accumulate
  return __builtin_amdgcn_wmma_f32_16x16x32_bf16(false, a, false, b, (short)0, c,
                                                 false, false);
}

__device__ __forceinline__ v8f v8f_zero() {
  v8f z = {0.f,0.f,0.f,0.f,0.f,0.f,0.f,0.f};
  return z;
}

// ---------------------------------------------------------------- convert ---
__global__ void mha_cvt_bf16(const float* __restrict__ in,
                             ushort_t* __restrict__ out, int n) {
  int i = (blockIdx.x * blockDim.x + threadIdx.x) * 4;
  if (i + 3 < n) {
    float4 f = *(const float4*)(in + i);
    uint2 p;
    p.x = pk2bf(f.x, f.y);
    p.y = pk2bf(f.z, f.w);
    *(uint2*)(out + i) = p;
  }
}

// out[b][c][r] = bf16(in[b][r][c])  -- pre-transpose weights so WMMA B
// fragments ([N][K] memory) are contiguous 32B per-lane loads.
__global__ void mha_transpose_bf16(const float* __restrict__ in,
                                   ushort_t* __restrict__ out, int R, int C) {
  long long t  = (long long)blockIdx.x * blockDim.x + threadIdx.x;
  long long rc = (long long)R * C;
  int b   = (int)(t / rc);
  int rem = (int)(t - (long long)b * rc);
  int c   = rem / R;
  int r   = rem - c * R;
  out[t] = f2bf_bits(in[(long long)b * rc + (long long)r * C + c]);
}

// ------------------------------------------------------------- QKV GEMMs ---
// Per block: one (b,h), 128 s-rows. Per wave: 16 rows x 64 cols for Q, K, V.
__global__ __launch_bounds__(256) void mha_qkv_proj(
    const ushort_t* __restrict__ Xb,   // [B][S][M] bf16
    const ushort_t* __restrict__ WqT,  // [H][D][M] bf16 (transposed)
    const ushort_t* __restrict__ WkT,
    const ushort_t* __restrict__ WvT,
    ushort_t* __restrict__ Qb,         // [B][H][S][D]
    ushort_t* __restrict__ Kb,         // [B][H][S][D]
    ushort_t* __restrict__ Vt) {       // [B][H][D][S] (transposed for AV)
  const int lane = threadIdx.x & 31;
  const int w    = threadIdx.x >> 5;
  const int half = lane >> 4;
  const int nloc = lane & 15;
  const int bh   = blockIdx.y;
  const int b    = bh >> 4, h = bh & 15;
  const int s0   = blockIdx.x * 128 + w * 16;

  v8f accQ[4], accK[4], accV[4];
#pragma unroll
  for (int i = 0; i < 4; i++) { accQ[i] = v8f_zero(); accK[i] = v8f_zero(); accV[i] = v8f_zero(); }

  const ushort_t* xrow = Xb + ((long long)b * S_LEN + (s0 + nloc)) * DMODEL;
  const ushort_t* wq = WqT + (long long)h * DHEAD * DMODEL;
  const ushort_t* wk = WkT + (long long)h * DHEAD * DMODEL;
  const ushort_t* wv = WvT + (long long)h * DHEAD * DMODEL;

  for (int kc = 0; kc < DMODEL; kc += 32) {
    v8bf lo = *(const v8bf*)(xrow + kc + half * 16);
    v8bf hi = *(const v8bf*)(xrow + kc + half * 16 + 8);
    v16bf a = cat8(lo, hi);

    // Load the whole k-step's B fragments first (clause-friendly), then the
    // WMMA burst.
    v16bf bq[4], bk[4], bv[4];
#pragma unroll
    for (int nt = 0; nt < 4; nt++) {
      long long boff = (long long)(nt * 16 + nloc) * DMODEL + kc + half * 16;
      bq[nt] = *(const v16bf*)(wq + boff);
      bk[nt] = *(const v16bf*)(wk + boff);
      bv[nt] = *(const v16bf*)(wv + boff);
    }
#pragma unroll
    for (int nt = 0; nt < 4; nt++) {
      accQ[nt] = wmma_bf16(a, bq[nt], accQ[nt]);
      accK[nt] = wmma_bf16(a, bk[nt], accK[nt]);
      accV[nt] = wmma_bf16(a, bv[nt], accV[nt]);
    }
  }

  const long long base = (long long)bh * S_LEN;
#pragma unroll
  for (int nt = 0; nt < 4; nt++) {
#pragma unroll
    for (int i = 0; i < 8; i++) {
      int row = s0 + i + half * 8;
      int col = nt * 16 + nloc;
      Qb[(base + row) * DHEAD + col] = f2bf_bits(accQ[nt][i]);
      Kb[(base + row) * DHEAD + col] = f2bf_bits(accK[nt][i]);
    }
    // V epilogue: elements i are contiguous along s in the transposed layout
    // -> one 16B packed store per fragment instead of 8 scalar b16 stores.
    uint4 pv;
    pv.x = pk2bf(accV[nt][0], accV[nt][1]);
    pv.y = pk2bf(accV[nt][2], accV[nt][3]);
    pv.z = pk2bf(accV[nt][4], accV[nt][5]);
    pv.w = pk2bf(accV[nt][6], accV[nt][7]);
    int col = nt * 16 + nloc;
    *(uint4*)(Vt + ((long long)bh * DHEAD + col) * S_LEN + s0 + half * 8) = pv;
  }
}

// --------------------------------------------------------- flash attention --
// Per block: one (b,h), 128 q-rows; each wave owns 16 q-rows independently.
__global__ __launch_bounds__(256) void mha_flash_attn(
    const ushort_t* __restrict__ Qb,  // [B][H][S][D]
    const ushort_t* __restrict__ Kb,  // [B][H][S][D]
    const ushort_t* __restrict__ Vt,  // [B][H][D][S]
    ushort_t* __restrict__ Ob) {      // [B][S][H*D] concat layout
  __shared__ ushort_t pstage[8 * 16 * 32];  // per-wave P staging (C->A layout)

  const int lane = threadIdx.x & 31;
  const int w    = threadIdx.x >> 5;
  const int half = lane >> 4;
  const int nloc = lane & 15;
  const int bh   = blockIdx.y;
  const int b    = bh >> 4, h = bh & 15;
  const int q0   = blockIdx.x * 128 + w * 16;

  const long long qkbase = (long long)bh * S_LEN * DHEAD;
  const long long vbase  = (long long)bh * DHEAD * S_LEN;

  // Q fragments resident for the whole k loop (A layout, d_head = 2x32)
  const ushort_t* qrow = Qb + qkbase + (long long)(q0 + nloc) * DHEAD;
  v16bf aQ0 = cat8(*(const v8bf*)(qrow + 0  + half * 16),
                   *(const v8bf*)(qrow + 8  + half * 16));
  v16bf aQ1 = cat8(*(const v8bf*)(qrow + 32 + half * 16),
                   *(const v8bf*)(qrow + 40 + half * 16));

  v8f accO[4];
#pragma unroll
  for (int i = 0; i < 4; i++) accO[i] = v8f_zero();
  float m_i[8], l_i[8];
#pragma unroll
  for (int i = 0; i < 8; i++) { m_i[i] = -3.0e38f; l_i[i] = 0.f; }

  const int kend = q0 + 16;                 // causal upper bound (exclusive)
  for (int kt = 0; kt < kend; kt += 32) {
    // ---- scores: S(16x32) = Q(16x64) @ K^T ----
    const ushort_t* krow0 = Kb + qkbase + (long long)(kt + nloc) * DHEAD + half * 16;
    const ushort_t* krow1 = krow0 + 16 * DHEAD;
    v16bf bk00 = *(const v16bf*)(krow0);
    v16bf bk01 = *(const v16bf*)(krow0 + 32);
    v16bf bk10 = *(const v16bf*)(krow1);
    v16bf bk11 = *(const v16bf*)(krow1 + 32);
    v8f s0v = v8f_zero(), s1v = v8f_zero();
    s0v = wmma_bf16(aQ0, bk00, s0v);
    s0v = wmma_bf16(aQ1, bk01, s0v);
    s1v = wmma_bf16(aQ0, bk10, s1v);
    s1v = wmma_bf16(aQ1, bk11, s1v);

    if (kt + 32 < kend)
      __builtin_prefetch(Kb + qkbase + (long long)(kt + 32 + nloc) * DHEAD, 0, 1);

    // ---- scale, causal mask, online softmax (rows live in 16-lane halves) --
    float p0[8], p1[8];
#pragma unroll
    for (int i = 0; i < 8; i++) {
      int qr = q0 + i + half * 8;
      float x0 = s0v[i] * 0.125f;            // 1/sqrt(64)
      float x1 = s1v[i] * 0.125f;
      if (kt + nloc      > qr) x0 = -1e30f;
      if (kt + 16 + nloc > qr) x1 = -1e30f;
      float rmax = fmaxf(x0, x1);
#pragma unroll
      for (int off = 1; off < 16; off <<= 1)
        rmax = fmaxf(rmax, __shfl_xor(rmax, off, 32));
      float mnew  = fmaxf(m_i[i], rmax);
      float alpha = __expf(m_i[i] - mnew);
      m_i[i] = mnew;
      float e0 = __expf(x0 - mnew);
      float e1 = __expf(x1 - mnew);
      float rsum = e0 + e1;
#pragma unroll
      for (int off = 1; off < 16; off <<= 1)
        rsum += __shfl_xor(rsum, off, 32);
      l_i[i] = l_i[i] * alpha + rsum;
#pragma unroll
      for (int dt = 0; dt < 4; dt++) accO[dt][i] = accO[dt][i] * alpha;
      p0[i] = e0; p1[i] = e1;
    }

    // ---- P: C layout -> A layout via per-wave LDS slice ----
    ushort_t* pw = pstage + w * 512;
#pragma unroll
    for (int i = 0; i < 8; i++) {
      int row = i + half * 8;
      pw[row * 32 + nloc]      = f2bf_bits(p0[i]);
      pw[row * 32 + 16 + nloc] = f2bf_bits(p1[i]);
    }
    asm volatile("s_wait_dscnt 0" ::: "memory");   // CDNA5 split DS counter
    const ushort_t* pr = pw + nloc * 32 + half * 16;
    v16bf aP = cat8(*(const v8bf*)(pr), *(const v8bf*)(pr + 8));

    // ---- O += P(16x32) @ V(32x64), V stored transposed [d][s] ----
    const ushort_t* vrow = Vt + vbase + (long long)nloc * S_LEN + kt + half * 16;
    v16bf bv[4];
#pragma unroll
    for (int dt = 0; dt < 4; dt++)
      bv[dt] = *(const v16bf*)(vrow + (long long)dt * 16 * S_LEN);
#pragma unroll
    for (int dt = 0; dt < 4; dt++)
      accO[dt] = wmma_bf16(aP, bv[dt], accO[dt]);
  }

  // ---- normalize, write concat-head layout [b][s][h*64+d] ----
#pragma unroll
  for (int i = 0; i < 8; i++) {
    float inv = 1.0f / l_i[i];
    int srow = q0 + i + half * 8;
    ushort_t* orow = Ob + ((long long)b * S_LEN + srow) * DMODEL + h * DHEAD;
#pragma unroll
    for (int dt = 0; dt < 4; dt++)
      orow[dt * 16 + nloc] = f2bf_bits(accO[dt][i] * inv);
  }
}

// ----------------------------------------------------- output projection ---
__global__ __launch_bounds__(256) void mha_out_proj(
    const ushort_t* __restrict__ Ob,   // [B*S][M] bf16
    const ushort_t* __restrict__ WoT,  // [N][M]  bf16 (transposed)
    float* __restrict__ Y) {           // [B*S][N] f32
  const int lane = threadIdx.x & 31;
  const int w    = threadIdx.x >> 5;
  const int half = lane >> 4;
  const int nloc = lane & 15;
  const int r0   = blockIdx.x * 128 + w * 16;
  const int n0   = blockIdx.y * 64;

  v8f acc[4];
#pragma unroll
  for (int i = 0; i < 4; i++) acc[i] = v8f_zero();

  const ushort_t* arow = Ob + (long long)(r0 + nloc) * DMODEL;
  for (int mc = 0; mc < DMODEL; mc += 32) {
    v16bf a = cat8(*(const v8bf*)(arow + mc + half * 16),
                   *(const v8bf*)(arow + mc + half * 16 + 8));
    v16bf bw[4];
#pragma unroll
    for (int nt = 0; nt < 4; nt++)
      bw[nt] = *(const v16bf*)(WoT + (long long)(n0 + nt * 16 + nloc) * DMODEL +
                               mc + half * 16);
#pragma unroll
    for (int nt = 0; nt < 4; nt++)
      acc[nt] = wmma_bf16(a, bw[nt], acc[nt]);
  }
#pragma unroll
  for (int nt = 0; nt < 4; nt++)
#pragma unroll
    for (int i = 0; i < 8; i++)
      Y[(long long)(r0 + i + half * 8) * DMODEL + n0 + nt * 16 + nloc] = acc[nt][i];
}

// ------------------------------------------------------------------ host ---
extern "C" void kernel_launch(void* const* d_in, const int* in_sizes, int n_in,
                              void* d_out, int out_size, void* d_ws, size_t ws_size,
                              hipStream_t stream) {
  (void)in_sizes; (void)n_in; (void)out_size; (void)ws_size;

  const float* X  = (const float*)d_in[0];
  const float* Wq = (const float*)d_in[1];
  const float* Wk = (const float*)d_in[2];
  const float* Wv = (const float*)d_in[3];
  const float* Wo = (const float*)d_in[4];
  float*       Y  = (float*)d_out;

  const size_t nX  = (size_t)NBATCH * S_LEN * DMODEL;            // 4,194,304
  const size_t nW  = (size_t)NHEADS * DMODEL * DHEAD;            // 1,048,576
  const size_t nWo = (size_t)DMODEL * DMODEL;                    // 1,048,576
  const size_t nQ  = (size_t)NBATCH * NHEADS * S_LEN * DHEAD;    // 4,194,304

  ushort_t* Xb  = (ushort_t*)d_ws;
  ushort_t* WqT = Xb  + nX;
  ushort_t* WkT = WqT + nW;
  ushort_t* WvT = WkT + nW;
  ushort_t* WoT = WvT + nW;
  ushort_t* Qb  = WoT + nWo;
  ushort_t* Kb  = Qb  + nQ;
  ushort_t* Vt  = Kb  + nQ;
  ushort_t* Ob  = Vt  + nQ;   // total ~48 MB of workspace

  // 1) fp32 -> bf16 (weights pre-transposed for contiguous B-fragment loads)
  mha_cvt_bf16<<<(int)(nX / 1024), 256, 0, stream>>>(X, Xb, (int)nX);
  mha_transpose_bf16<<<(int)(nW / 256), 256, 0, stream>>>(Wq, WqT, DMODEL, DHEAD);
  mha_transpose_bf16<<<(int)(nW / 256), 256, 0, stream>>>(Wk, WkT, DMODEL, DHEAD);
  mha_transpose_bf16<<<(int)(nW / 256), 256, 0, stream>>>(Wv, WvT, DMODEL, DHEAD);
  mha_transpose_bf16<<<(int)(nWo / 256), 256, 0, stream>>>(Wo, WoT, DMODEL, DMODEL);

  // 2) QKV projections (V written transposed)
  mha_qkv_proj<<<dim3(S_LEN / 128, NBATCH * NHEADS), 256, 0, stream>>>(
      Xb, WqT, WkT, WvT, Qb, Kb, Vt);

  // 3) causal flash attention
  mha_flash_attn<<<dim3(S_LEN / 128, NBATCH * NHEADS), 256, 0, stream>>>(
      Qb, Kb, Vt, Ob);

  // 4) output projection -> f32
  mha_out_proj<<<dim3((NBATCH * S_LEN) / 128, DMODEL / 64), 256, 0, stream>>>(
      Ob, WoT, Y);
}
